// LocationAttention_55327768707522
// MI455X (gfx1250) — matrix operand order repaired and checked by reading, not compiled, not run on hardware
//
#include <hip/hip_runtime.h>
#include <hip/hip_bf16.h>
#include <stdint.h>

// ---- problem constants ----
#define ATTN_DIMc 512
#define ENC_DIMc  1024
#define DEC_DIMc  1024
#define CONV_DIMc 64
#define CONV_Kc   15
#define Bc        16
#define Tc        2048
#define SCALINGc  2.0f

typedef __attribute__((ext_vector_type(16))) __bf16 v16bf;
typedef __attribute__((ext_vector_type(8)))  float  v8f;

static __device__ __forceinline__ uint16_t f32_to_bf16(float f) {
  uint32_t u = __float_as_uint(f);
  uint32_t r = u + 0x7FFFu + ((u >> 16) & 1u);
  return (uint16_t)(r >> 16);
}

// ---------------- K1: convert W_enc / W_attn to bf16 ----------------
__global__ void convert_weights_kernel(const float* __restrict__ W_enc,
                                       const float* __restrict__ W_attn,
                                       uint16_t* __restrict__ wenc_bf,
                                       uint16_t* __restrict__ wattn_bf) {
  int i = blockIdx.x * 256 + threadIdx.x;
  const int n1 = ATTN_DIMc * ENC_DIMc;   // 524288
  const int n2 = ATTN_DIMc * CONV_DIMc;  // 32768
  if (i < n1) wenc_bf[i]  = f32_to_bf16(W_enc[i]);
  if (i < n2) wattn_bf[i] = f32_to_bf16(W_attn[i]);
}

// ---------------- K2: bias[b,a] = dec_h[b]·W_dec[a] + b_enc[a] ----------------
__global__ void bias_kernel(const float* __restrict__ dec_h,
                            const float* __restrict__ W_dec,
                            const float* __restrict__ b_enc,
                            float* __restrict__ bias) {
  int i = blockIdx.x * 256 + threadIdx.x;   // 0..8191
  int b = i & (Bc - 1);
  int a = i >> 4;                           // 0..511
  const float* dh = dec_h + (size_t)b * DEC_DIMc;
  const float* wd = W_dec + (size_t)a * DEC_DIMc;
  float acc = b_enc[a];
  #pragma unroll 4
  for (int d = 0; d < DEC_DIMc; ++d) acc += dh[d] * wd[d];
  bias[b * ATTN_DIMc + a] = acc;
}

// ---------------- K3: conv1d (cross-correlation, same padding) -> bf16 [B,T,64]
__global__ void conv_kernel(const float* __restrict__ attn_state,
                            const float* __restrict__ W_conv,
                            uint16_t* __restrict__ conv_bf) {
  __shared__ float wconv[CONV_DIMc * 2 * 31];
  int tid = threadIdx.x;
  for (int i = tid; i < CONV_DIMc * 2 * 31; i += 256) wconv[i] = W_conv[i];
  __syncthreads();
  int b = blockIdx.y;
  int t = blockIdx.x * 256 + tid;
  float win[2][31];
  #pragma unroll
  for (int s = 0; s < 2; ++s)
    #pragma unroll
    for (int k = 0; k < 31; ++k) {
      int tt = t + k - CONV_Kc;
      win[s][k] = (tt >= 0 && tt < Tc) ? attn_state[((size_t)b * 2 + s) * Tc + tt] : 0.f;
    }
  uint16_t* outp = conv_bf + ((size_t)b * Tc + t) * CONV_DIMc;
  for (int c = 0; c < CONV_DIMc; ++c) {
    const float* w = wconv + c * 62;
    float acc = 0.f;
    #pragma unroll
    for (int s = 0; s < 2; ++s)
      #pragma unroll
      for (int k = 0; k < 31; ++k) acc += win[s][k] * w[s * 31 + k];
    outp[c] = f32_to_bf16(acc);
  }
}

// ---------------- K4: main WMMA score kernel ----------------
#define M_TILE 64
#define ENC_PITCH 1032   // 1024 + 8 elems -> 2064B row pitch, bank-conflict-free b128
#define CONV_PITCH 72    // 64 + 8

union FragU { struct { uint4 lo, hi; } q; v16bf v; };

__global__ __launch_bounds__(128)
void score_kernel(const float* __restrict__ enc,
                  const uint16_t* __restrict__ wenc_bf,
                  const uint16_t* __restrict__ wattn_bf,
                  const uint16_t* __restrict__ conv_bf,
                  const float* __restrict__ bias,
                  const float* __restrict__ W_out,
                  float* __restrict__ scores) {
  extern __shared__ uint16_t smem[];
  uint16_t* encT  = smem;                          // [64][ENC_PITCH]
  uint16_t* convT = smem + M_TILE * ENC_PITCH;     // [64][CONV_PITCH]

  const int tid = threadIdx.x;
  const int b   = blockIdx.y;
  const int t0  = blockIdx.x * M_TILE;

  // stage encoder tile (f32 -> bf16), coalesced
  const float* encp = enc + ((size_t)b * Tc + t0) * ENC_DIMc;
  for (int i = tid; i < M_TILE * ENC_DIMc; i += 128) {
    int r = i >> 10, c = i & 1023;
    encT[r * ENC_PITCH + c] = f32_to_bf16(encp[(size_t)r * ENC_DIMc + c]);
  }
  // stage conv tile (already bf16)
  const uint16_t* convp = conv_bf + ((size_t)b * Tc + t0) * CONV_DIMc;
  for (int i = tid; i < M_TILE * CONV_DIMc; i += 128) {
    int r = i >> 6, c = i & 63;
    convT[r * CONV_PITCH + c] = convp[r * CONV_DIMc + c];
  }
  __syncthreads();

  const int wv   = tid >> 5;      // wave id: owns rows 16*wv .. 16*wv+15
  const int lane = tid & 31;
  const int hf   = lane >> 4;     // K-half for A/B fragments
  const int n    = lane & 15;     // A row within tile / B (and C) column

  // per-lane A fragment bases (ISA 16-bit A 16x32 layout):
  //   VGPR0-3: K = hf*8 + 0..7 ; VGPR4-7: K = 16 + hf*8 + 0..7
  const uint16_t* a0p  = encT  + ((wv << 4) + n) * ENC_PITCH + hf * 8;
  const uint16_t* a1p  = a0p + 16;
  const uint16_t* ca0p = convT + ((wv << 4) + n) * CONV_PITCH + hf * 8;
  const uint16_t* ca1p = ca0p + 16;

  float scoreAcc[8];
  #pragma unroll
  for (int r = 0; r < 8; ++r) scoreAcc[r] = 0.f;

  for (int chunk = 0; chunk < 32; ++chunk) {
    const int acol = (chunk << 4) + n;           // this lane's B/C column (a index)
    // B fragment (32x16): same K order as A: K = hf*8 + 0..7, then 16 + hf*8 + 0..7
    const uint16_t* bp  = wenc_bf  + (size_t)acol * ENC_DIMc  + hf * 8;
    const uint16_t* cbp = wattn_bf + (size_t)acol * CONV_DIMc + hf * 8;

    v8f acc = {};
    #pragma unroll 8
    for (int kk = 0; kk < 32; ++kk) {           // K = 1024 over encoder dim
      FragU af, bf2;
      af.q.lo  = *(const uint4*)(a0p + kk * 32);
      af.q.hi  = *(const uint4*)(a1p + kk * 32);
      bf2.q.lo = *(const uint4*)(bp  + kk * 32);
      bf2.q.hi = *(const uint4*)(bp  + kk * 32 + 16);
      acc = __builtin_amdgcn_wmma_f32_16x16x32_bf16(
          false, af.v, false, bf2.v, (short)0, acc, false, false);
    }
    #pragma unroll
    for (int kk = 0; kk < 2; ++kk) {            // K = 64 over conv channels
      FragU af, bf2;
      af.q.lo  = *(const uint4*)(ca0p + kk * 32);
      af.q.hi  = *(const uint4*)(ca1p + kk * 32);
      bf2.q.lo = *(const uint4*)(cbp  + kk * 32);
      bf2.q.hi = *(const uint4*)(cbp  + kk * 32 + 16);
      acc = __builtin_amdgcn_wmma_f32_16x16x32_bf16(
          false, af.v, false, bf2.v, (short)0, acc, false, false);
    }
    const float bv = bias[(b << 9) + acol];
    const float wo = W_out[acol];
    #pragma unroll
    for (int r = 0; r < 8; ++r)                 // C layout: lane holds (M=r+8*hf, N=n)
      scoreAcc[r] += wo * tanhf(acc[r] + bv);
  }

  // reduce over the 16 columns (lanes within same half; xor<16 preserves hf bit)
  #pragma unroll
  for (int r = 0; r < 8; ++r) {
    float v = scoreAcc[r];
    v += __shfl_xor(v, 1, 32);
    v += __shfl_xor(v, 2, 32);
    v += __shfl_xor(v, 4, 32);
    v += __shfl_xor(v, 8, 32);
    scoreAcc[r] = v;
  }
  if (n == 0) {
    #pragma unroll
    for (int r = 0; r < 8; ++r) {
      int row = (wv << 4) + (hf << 3) + r;      // C rows: r + 8*hf within wave tile
      scores[(b << 11) + t0 + row] = scoreAcc[r];
    }
  }
}

// ---------------- K5: per-batch masked softmax over T ----------------
__global__ void softmax_kernel(const float* __restrict__ scores,
                               const uint8_t* __restrict__ mask,
                               float* __restrict__ w_out) {
  __shared__ float red[256];
  int b = blockIdx.x, tid = threadIdx.x;
  const float* s = scores + (size_t)b * Tc;
  const uint8_t* m = mask + (size_t)b * Tc;
  float vals[8];
  float mx = -INFINITY;
  #pragma unroll
  for (int i = 0; i < 8; ++i) {
    int t = tid + i * 256;
    float v = m[t] ? -INFINITY : s[t];
    vals[i] = v;
    mx = fmaxf(mx, v);
  }
  red[tid] = mx; __syncthreads();
  for (int off = 128; off > 0; off >>= 1) {
    if (tid < off) red[tid] = fmaxf(red[tid], red[tid + off]);
    __syncthreads();
  }
  mx = red[0]; __syncthreads();
  float sum = 0.f;
  #pragma unroll
  for (int i = 0; i < 8; ++i) {
    float e = (vals[i] == -INFINITY) ? 0.f : __expf(SCALINGc * (vals[i] - mx));
    vals[i] = e; sum += e;
  }
  red[tid] = sum; __syncthreads();
  for (int off = 128; off > 0; off >>= 1) {
    if (tid < off) red[tid] += red[tid + off];
    __syncthreads();
  }
  float inv = 1.f / red[0];
  #pragma unroll
  for (int i = 0; i < 8; ++i) w_out[(size_t)b * Tc + tid + i * 256] = vals[i] * inv;
}

// ---------------- K6: c[b,e] = sum_t enc[b,t,e] * w[b,t] ----------------
__global__ void context_kernel(const float* __restrict__ enc,
                               const float* __restrict__ w,
                               float* __restrict__ c_out) {
  int b = blockIdx.y;
  int e = blockIdx.x * 256 + threadIdx.x;
  const float* ep = enc + (size_t)b * Tc * ENC_DIMc + e;
  const float* wp = w + (size_t)b * Tc;
  float acc = 0.f;
  #pragma unroll 4
  for (int t = 0; t < Tc; ++t) acc += ep[(size_t)t * ENC_DIMc] * wp[t];
  c_out[b * ENC_DIMc + e] = acc;
}

// ---------------- launch ----------------
extern "C" void kernel_launch(void* const* d_in, const int* in_sizes, int n_in,
                              void* d_out, int out_size, void* d_ws, size_t ws_size,
                              hipStream_t stream) {
  const float*   enc        = (const float*)d_in[0];
  const uint8_t* mask       = (const uint8_t*)d_in[1];   // bool array
  const float*   dec_h      = (const float*)d_in[2];
  const float*   attn_state = (const float*)d_in[3];
  const float*   W_enc      = (const float*)d_in[4];
  const float*   b_enc      = (const float*)d_in[5];
  const float*   W_dec      = (const float*)d_in[6];
  const float*   W_attn     = (const float*)d_in[7];
  const float*   W_conv     = (const float*)d_in[8];
  const float*   W_out      = (const float*)d_in[9];
  // d_in[10] = b_out: constant shift before softmax -> no effect on outputs.

  char* ws = (char*)d_ws;
  uint16_t* wenc_bf  = (uint16_t*)(ws);                 // 1,048,576 B
  uint16_t* wattn_bf = (uint16_t*)(ws + 1048576);       //    65,536 B
  uint16_t* conv_bf  = (uint16_t*)(ws + 1114112);       // 4,194,304 B
  float*    bias     = (float*)   (ws + 5308416);       //    32,768 B
  float*    scores   = (float*)   (ws + 5341184);       //   131,072 B

  float* out   = (float*)d_out;
  float* c_out = out;                  // [16,1024]
  float* w_out = out + Bc * ENC_DIMc;  // [16,2048]

  convert_weights_kernel<<<2048, 256, 0, stream>>>(W_enc, W_attn, wenc_bf, wattn_bf);
  bias_kernel<<<32, 256, 0, stream>>>(dec_h, W_dec, b_enc, bias);
  conv_kernel<<<dim3(Tc / 256, Bc), 256, 0, stream>>>(attn_state, W_conv, conv_bf);

  size_t smem = (size_t)(M_TILE * ENC_PITCH + M_TILE * CONV_PITCH) * sizeof(uint16_t);
  (void)hipFuncSetAttribute((const void*)score_kernel,
                            hipFuncAttributeMaxDynamicSharedMemorySize, (int)smem);
  score_kernel<<<dim3(Tc / M_TILE, Bc), 128, smem, stream>>>(
      enc, wenc_bf, wattn_bf, conv_bf, bias, W_out, scores);

  softmax_kernel<<<Bc, 256, 0, stream>>>(scores, mask, w_out);
  context_kernel<<<dim3(ENC_DIMc / 256, Bc), 256, 0, stream>>>(enc, w_out, c_out);
}
